// DotProductAttention_57853209477398
// MI455X (gfx1250) — matrix operand order repaired and checked by reading, not compiled, not run on hardware
//
#include <hip/hip_runtime.h>
#include <hip/hip_bf16.h>

typedef __attribute__((ext_vector_type(16))) __bf16 v16bf;
typedef __attribute__((ext_vector_type(8)))  __bf16 v8bf;
typedef __attribute__((ext_vector_type(8)))  float  v8f;

#define B_   32
#define LQ_  128
#define FQ_  1024
#define LK_  512
#define FK_  1024
#define LV_  512
#define FV_  1024
#define DP_  128
#define RSQRT_D 0.08838834764831843f   // 1/sqrt(128)

#define WMMA_BF16(a, b, c) \
  __builtin_amdgcn_wmma_f32_16x16x32_bf16(false, (a), false, (b), (short)0, (c), false, false)

// ---- bf16 helpers (bit-exact RNE, storage-only use of __bf16) ----
__device__ __forceinline__ __bf16 f2bf(float f) {
  union { float f; unsigned u; } v; v.f = f;
  unsigned r = v.u + 0x7FFFu + ((v.u >> 16) & 1u);
  unsigned short h = (unsigned short)(r >> 16);
  return __builtin_bit_cast(__bf16, h);
}
__device__ __forceinline__ float bf2f(__bf16 b) {
  unsigned short h = __builtin_bit_cast(unsigned short, b);
  union { unsigned u; float f; } v; v.u = ((unsigned)h) << 16;
  return v.f;
}

// ---- packed WMMA fragment loads (K contiguous in memory) ----
// A 16x32 bf16 (ISA 7.12.2): lane half hi=0 -> K 0-7 & 16-23; hi=1 -> K 8-15 & 24-31.
__device__ __forceinline__ v16bf ld_a(const __bf16* rowp, int kk, int hi) {
  v8bf lo = *(const v8bf*)(rowp + kk + hi * 8);
  v8bf hv = *(const v8bf*)(rowp + kk + 16 + hi * 8);
  return __builtin_shufflevector(lo, hv, 0,1,2,3,4,5,6,7,8,9,10,11,12,13,14,15);
}
// B 32x16 bf16: lane = column, lane half hi=0 -> K 0-7 & 16-23; hi=1 -> K 8-15 & 24-31.
__device__ __forceinline__ v16bf ld_b(const __bf16* colp, int kk, int hi) {
  v8bf lo = *(const v8bf*)(colp + kk + hi * 8);
  v8bf hv = *(const v8bf*)(colp + kk + 16 + hi * 8);
  return __builtin_shufflevector(lo, hv, 0,1,2,3,4,5,6,7,8,9,10,11,12,13,14,15);
}

// ============ fp32->bf16 transpose (LDS-tiled, coalesced both ways) ============
// src [z][R][C] fp32 -> dst [z][C][R] bf16 ; grid (C/32, R/32, Z), block (32,8)
__global__ void transpose_cvt_kernel(const float* __restrict__ src,
                                     __bf16* __restrict__ dst, int R, int C) {
  __shared__ __bf16 tile[32][34];
  const float* s = src + (size_t)blockIdx.z * R * C;
  __bf16*      d = dst + (size_t)blockIdx.z * R * C;
  const int x0 = blockIdx.x * 32, y0 = blockIdx.y * 32;
  const int tx = threadIdx.x, ty = threadIdx.y;
#pragma unroll
  for (int r = ty; r < 32; r += 8)
    tile[r][tx] = f2bf(s[(size_t)(y0 + r) * C + x0 + tx]);
  __syncthreads();
#pragma unroll
  for (int r = ty; r < 32; r += 8)
    d[(size_t)(x0 + r) * R + y0 + tx] = tile[tx][r];
}

// ============ fp32->bf16 straight convert, 8 elements/thread ============
__global__ void cvt_kernel(const float* __restrict__ src, __bf16* __restrict__ dst) {
  const size_t i = (size_t)blockIdx.x * blockDim.x + threadIdx.x;
  const float4* s = (const float4*)src;
  float4 a = s[2 * i], b = s[2 * i + 1];
  v8bf o;
  o[0]=f2bf(a.x); o[1]=f2bf(a.y); o[2]=f2bf(a.z); o[3]=f2bf(a.w);
  o[4]=f2bf(b.x); o[5]=f2bf(b.y); o[6]=f2bf(b.z); o[7]=f2bf(b.w);
  ((v8bf*)dst)[i] = o;
}

// ============ GEMM 1: kproj[b,f,d] = sum_l keysT[b,f,l] * W1[d,l] ============
// Software-pipelined: loads for kk+32 issued before WMMAs for kk.
__global__ void kproj_kernel(const __bf16* __restrict__ keysT,
                             const __bf16* __restrict__ w1bf,
                             __bf16* __restrict__ kproj) {
  const int wave = blockIdx.x * (blockDim.x >> 5) + (threadIdx.x >> 5);
  const int lane = threadIdx.x & 31;
  const int b = wave >> 7, rem = wave & 127;       // 64 Mtiles * 2 Ngroups
  const int tm = rem >> 1, n0 = (rem & 1) * 64;
  const int hi = lane >> 4, idx = lane & 15;
  const __bf16* rowp = keysT + (size_t)b * FK_ * LK_ + (size_t)(tm * 16 + idx) * LK_;
  const __bf16* colp0 = w1bf + (size_t)(n0 + idx) * LK_;
  const __bf16* colp1 = colp0 + 16 * LK_;
  const __bf16* colp2 = colp0 + 32 * LK_;
  const __bf16* colp3 = colp0 + 48 * LK_;
  v8f c0 = {}, c1 = {}, c2 = {}, c3 = {};
  v16bf a  = ld_a(rowp, 0, hi);
  v16bf b0 = ld_b(colp0, 0, hi), b1 = ld_b(colp1, 0, hi);
  v16bf b2 = ld_b(colp2, 0, hi), b3 = ld_b(colp3, 0, hi);
#pragma unroll 2
  for (int kk = 32; kk < LK_; kk += 32) {
    v16bf an  = ld_a(rowp, kk, hi);
    v16bf bn0 = ld_b(colp0, kk, hi), bn1 = ld_b(colp1, kk, hi);
    v16bf bn2 = ld_b(colp2, kk, hi), bn3 = ld_b(colp3, kk, hi);
    c0 = WMMA_BF16(a, b0, c0); c1 = WMMA_BF16(a, b1, c1);
    c2 = WMMA_BF16(a, b2, c2); c3 = WMMA_BF16(a, b3, c3);
    a = an; b0 = bn0; b1 = bn1; b2 = bn2; b3 = bn3;
  }
  c0 = WMMA_BF16(a, b0, c0); c1 = WMMA_BF16(a, b1, c1);
  c2 = WMMA_BF16(a, b2, c2); c3 = WMMA_BF16(a, b3, c3);

  __bf16* kp = kproj + (size_t)b * FK_ * DP_;
#pragma unroll
  for (int r = 0; r < 8; ++r) {
    const size_t ro = (size_t)(tm * 16 + hi * 8 + r) * DP_ + n0 + idx;
    kp[ro + 0]  = f2bf(c0[r]);
    kp[ro + 16] = f2bf(c1[r]);
    kp[ro + 32] = f2bf(c2[r]);
    kp[ro + 48] = f2bf(c3[r]);
  }
}

// ============ GEMM 2: scores[b,i,j] = rsqrt(128) * sum_d qT[b,i,d] * kproj[b,j,d] ============
__global__ void scores_kernel(const __bf16* __restrict__ qT,
                              const __bf16* __restrict__ kproj,
                              __bf16* __restrict__ scores) {
  const int wave = blockIdx.x * (blockDim.x >> 5) + (threadIdx.x >> 5);
  const int lane = threadIdx.x & 31;
  const int b = wave >> 10, rem = wave & 1023;     // 64 Mtiles * 16 Ngroups
  const int tm = rem >> 4, n0 = (rem & 15) * 64;
  const int hi = lane >> 4, idx = lane & 15;
  const __bf16* rowp = qT + (size_t)b * FQ_ * DP_ + (size_t)(tm * 16 + idx) * DP_;
  const __bf16* colp0 = kproj + (size_t)b * FK_ * DP_ + (size_t)(n0 + idx) * DP_;
  const __bf16* colp1 = colp0 + 16 * DP_;
  const __bf16* colp2 = colp0 + 32 * DP_;
  const __bf16* colp3 = colp0 + 48 * DP_;
  v8f c0 = {}, c1 = {}, c2 = {}, c3 = {};
  v16bf a  = ld_a(rowp, 0, hi);
  v16bf b0 = ld_b(colp0, 0, hi), b1 = ld_b(colp1, 0, hi);
  v16bf b2 = ld_b(colp2, 0, hi), b3 = ld_b(colp3, 0, hi);
#pragma unroll
  for (int kk = 32; kk < DP_; kk += 32) {
    v16bf an  = ld_a(rowp, kk, hi);
    v16bf bn0 = ld_b(colp0, kk, hi), bn1 = ld_b(colp1, kk, hi);
    v16bf bn2 = ld_b(colp2, kk, hi), bn3 = ld_b(colp3, kk, hi);
    c0 = WMMA_BF16(a, b0, c0); c1 = WMMA_BF16(a, b1, c1);
    c2 = WMMA_BF16(a, b2, c2); c3 = WMMA_BF16(a, b3, c3);
    a = an; b0 = bn0; b1 = bn1; b2 = bn2; b3 = bn3;
  }
  c0 = WMMA_BF16(a, b0, c0); c1 = WMMA_BF16(a, b1, c1);
  c2 = WMMA_BF16(a, b2, c2); c3 = WMMA_BF16(a, b3, c3);

  __bf16* sB = scores + (size_t)b * FQ_ * FK_;
#pragma unroll
  for (int r = 0; r < 8; ++r) {
    const size_t ro = (size_t)(tm * 16 + hi * 8 + r) * FK_ + n0 + idx;
    sB[ro + 0]  = f2bf(c0[r] * RSQRT_D);
    sB[ro + 16] = f2bf(c1[r] * RSQRT_D);
    sB[ro + 32] = f2bf(c2[r] * RSQRT_D);
    sB[ro + 48] = f2bf(c3[r] * RSQRT_D);
  }
}

// ============ Softmax stats over axis i: m_j, 1/Z_j per column j ============
__global__ void colstats_kernel(const __bf16* __restrict__ scores,
                                float* __restrict__ stats) {
  __shared__ float red[8][32];
  const int b = blockIdx.x >> 5;
  const int j = (blockIdx.x & 31) * 32 + threadIdx.x;
  const int ty = threadIdx.y, tx = threadIdx.x;
  const __bf16* s = scores + (size_t)b * FQ_ * FK_ + j;

  float m = -3.402823466e38f;
  for (int i = ty; i < FQ_; i += 8)
    m = fmaxf(m, bf2f(s[(size_t)i * FK_]));
  red[ty][tx] = m;
  __syncthreads();
  if (ty == 0) {
#pragma unroll
    for (int r = 1; r < 8; ++r) m = fmaxf(m, red[r][tx]);
    red[0][tx] = m;
  }
  __syncthreads();
  m = red[0][tx];

  float z = 0.0f;
  for (int i = ty; i < FQ_; i += 8)
    z += __expf(bf2f(s[(size_t)i * FK_]) - m);
  __syncthreads();
  red[ty][tx] = z;
  __syncthreads();
  if (ty == 0) {
#pragma unroll
    for (int r = 1; r < 8; ++r) z += red[r][tx];
    stats[2 * (b * 1024 + j) + 0] = m;
    stats[2 * (b * 1024 + j) + 1] = 1.0f / z;
  }
}

// ============ attn = exp(s - m_j) * (1/Z_j), bf16 in place, 8-wide ============
__global__ void normalize_kernel(__bf16* __restrict__ scores,
                                 const float* __restrict__ stats) {
  const size_t t = (size_t)blockIdx.x * blockDim.x + threadIdx.x;
  v8bf* p = (v8bf*)scores;
  v8bf sv = p[t];
  const unsigned base = (unsigned)(t * 8);
  const unsigned j0 = base & 1023u;
  const unsigned b = base >> 20;
  const float* st = stats + 2u * ((b << 10) | j0);
  v8bf o;
#pragma unroll
  for (int e = 0; e < 8; ++e)
    o[e] = f2bf(__expf(bf2f(sv[e]) - st[2 * e]) * st[2 * e + 1]);
  p[t] = o;
}

// ============ GEMM 3: out[b,i,l] = sum_j attn[b,i,j] * values[b,l,j]  (fp32 out) ============
__global__ void out_kernel(const __bf16* __restrict__ attn,
                           const __bf16* __restrict__ vbf,
                           float* __restrict__ out) {
  const int wave = blockIdx.x * (blockDim.x >> 5) + (threadIdx.x >> 5);
  const int lane = threadIdx.x & 31;
  const int b = wave >> 9, rem = wave & 511;       // 64 Mtiles * 8 Ngroups
  const int tm = rem >> 3, n0 = (rem & 7) * 64;
  const int hi = lane >> 4, idx = lane & 15;
  const __bf16* rowp = attn + (size_t)b * FQ_ * FK_ + (size_t)(tm * 16 + idx) * FK_;
  const __bf16* colp0 = vbf + (size_t)b * LV_ * FV_ + (size_t)(n0 + idx) * FV_;
  const __bf16* colp1 = colp0 + 16 * FV_;
  const __bf16* colp2 = colp0 + 32 * FV_;
  const __bf16* colp3 = colp0 + 48 * FV_;
  v8f c0 = {}, c1 = {}, c2 = {}, c3 = {};
  v16bf a  = ld_a(rowp, 0, hi);
  v16bf b0 = ld_b(colp0, 0, hi), b1 = ld_b(colp1, 0, hi);
  v16bf b2 = ld_b(colp2, 0, hi), b3 = ld_b(colp3, 0, hi);
#pragma unroll 2
  for (int kk = 32; kk < FK_; kk += 32) {
    v16bf an  = ld_a(rowp, kk, hi);
    v16bf bn0 = ld_b(colp0, kk, hi), bn1 = ld_b(colp1, kk, hi);
    v16bf bn2 = ld_b(colp2, kk, hi), bn3 = ld_b(colp3, kk, hi);
    c0 = WMMA_BF16(a, b0, c0); c1 = WMMA_BF16(a, b1, c1);
    c2 = WMMA_BF16(a, b2, c2); c3 = WMMA_BF16(a, b3, c3);
    a = an; b0 = bn0; b1 = bn1; b2 = bn2; b3 = bn3;
  }
  c0 = WMMA_BF16(a, b0, c0); c1 = WMMA_BF16(a, b1, c1);
  c2 = WMMA_BF16(a, b2, c2); c3 = WMMA_BF16(a, b3, c3);

  float* oB = out + (size_t)b * FQ_ * LV_;
#pragma unroll
  for (int r = 0; r < 8; ++r) {
    const size_t ro = (size_t)(tm * 16 + hi * 8 + r) * LV_ + n0 + idx;
    oB[ro + 0]  = c0[r];
    oB[ro + 16] = c1[r];
    oB[ro + 32] = c2[r];
    oB[ro + 48] = c3[r];
  }
}

extern "C" void kernel_launch(void* const* d_in, const int* in_sizes, int n_in,
                              void* d_out, int out_size, void* d_ws, size_t ws_size,
                              hipStream_t stream) {
  const float* queries = (const float*)d_in[0];   // [32,128,1024]
  const float* keys    = (const float*)d_in[1];   // [32,512,1024]
  const float* values  = (const float*)d_in[2];   // [32,512,1024]
  const float* W1      = (const float*)d_in[3];   // [128,512]
  float* out = (float*)d_out;                     // [32,1024,512]

  // Workspace layout (scores aliases keysT, which is dead after GEMM1):
  char* ws = (char*)d_ws;
  __bf16* qT     = (__bf16*)(ws);                          //  8 MB  [b][i][d]
  __bf16* vbf    = (__bf16*)(ws + (8ull   << 20));         // 32 MB  [b][l][j]
  __bf16* kproj  = (__bf16*)(ws + (40ull  << 20));         //  8 MB  [b][j][d]
  __bf16* keysT  = (__bf16*)(ws + (48ull  << 20));         // 32 MB  [b][f][l]
  __bf16* scores = (__bf16*)(ws + (48ull  << 20));         // 64 MB  (aliases keysT)
  float*  stats  = (float*)(ws + (112ull << 20));          // 256 KB
  __bf16* w1bf   = (__bf16*)(ws + (112ull << 20) + (512ull << 10)); // 128 KB

  transpose_cvt_kernel<<<dim3(FK_/32, LK_/32, B_), dim3(32, 8), 0, stream>>>(keys, keysT, LK_, FK_);
  transpose_cvt_kernel<<<dim3(FQ_/32, LQ_/32, B_), dim3(32, 8), 0, stream>>>(queries, qT, LQ_, FQ_);
  cvt_kernel<<<(DP_ * LK_ / 8) / 256, 256, 0, stream>>>(W1, w1bf);
  cvt_kernel<<<(B_ * LV_ * FV_ / 8) / 256, 256, 0, stream>>>(values, vbf);

  kproj_kernel<<<4096 / 4, 128, 0, stream>>>(keysT, w1bf, kproj);
  scores_kernel<<<32768 / 4, 128, 0, stream>>>(qT, kproj, scores);
  colstats_kernel<<<B_ * 32, dim3(32, 8), 0, stream>>>(scores, stats);
  normalize_kernel<<<(33554432 / 8) / 256, 256, 0, stream>>>(scores, stats);
  out_kernel<<<16384 / 4, 128, 0, stream>>>(scores, vbf, out);
}
